// GPTLanguageModel_72816875536424
// MI455X (gfx1250) — matrix operand order repaired and checked by reading, not compiled, not run on hardware
//
#include <hip/hip_runtime.h>

// ---------------------------------------------------------------------------
// Types for gfx1250 WMMA (wave32): bf16 16x16x32, f32 accumulate.
// ---------------------------------------------------------------------------
typedef unsigned int   u32;
typedef unsigned short u16;
typedef __bf16 v16bf __attribute__((ext_vector_type(16)));
typedef float  v8f   __attribute__((ext_vector_type(8)));
typedef u32    u32x4 __attribute__((ext_vector_type(4)));
typedef float  f32x2 __attribute__((ext_vector_type(2)));

#define Tq    1024   // sequence length (== S)
#define Dm    1024   // model dim
#define Hh    16     // heads
#define HDh   64     // head dim
#define Ll    8      // layers
#define Mrows 2048   // B * T
#define D4    4096   // 4 * D
#define NV2   64000  // 2 * V

union Frag {
  u32x4 q[2];
  v16bf v;
  u16   s[16];
};

__device__ __forceinline__ u16 f2bf(float f) {
  // round-to-nearest-even fp32 -> bf16
  u32 u = __float_as_uint(f);
  u32 r = u + 0x7FFFu + ((u >> 16) & 1u);
  return (u16)(r >> 16);
}

__device__ __forceinline__ v8f wmma_bf16(v16bf a, v16bf b, v8f c) {
  // v_wmma_f32_16x16x32_bf16  (neg_a, A, neg_b, B, c_mod, C, reuse_a, reuse_b)
  return __builtin_amdgcn_wmma_f32_16x16x32_bf16(false, a, false, b, (short)0, c,
                                                 false, false);
}

// Pyramid mask (block_size = 1024, h = 512), T == S so lo == 0.
// masked == true means set score to -inf.
__device__ __forceinline__ bool is_masked(int t, int s) {
  if (s < 512) return s < 511 - (t >> 1);
  return (t == 0) || ((s - 512) > ((t - 1) >> 1));
}

// ---------------------------------------------------------------------------
// Embedding: x[b*T+t, :] = tok_emb[idx[b,t], :] + pos_emb[t, :]
// ---------------------------------------------------------------------------
__global__ void embed_kernel(const int* __restrict__ idx,
                             const float* __restrict__ tok,
                             const float* __restrict__ pos,
                             float* __restrict__ x) {
  const int row = blockIdx.x;           // b*T + t
  const int t   = row & (Tq - 1);
  const int tok_id = idx[row];
  const float* te = tok + (size_t)tok_id * Dm;
  const float* pe = pos + (size_t)t * Dm;
  float* xr = x + (size_t)row * Dm;
#pragma unroll
  for (int i = 0; i < 4; i++) {
    int c = threadIdx.x + i * 256;
    xr[c] = te[c] + pe[c];
  }
}

// ---------------------------------------------------------------------------
// LayerNorm (row of 1024) -> bf16 output
// ---------------------------------------------------------------------------
__global__ void layernorm_to_bf16(const float* __restrict__ x,
                                  const float* __restrict__ g,
                                  const float* __restrict__ b,
                                  u16* __restrict__ out) {
  __shared__ float r1[256];
  __shared__ float r2[256];
  const int row = blockIdx.x;
  const int tid = threadIdx.x;
  const float* xr = x + (size_t)row * Dm;
  float v[4], s = 0.f, s2 = 0.f;
#pragma unroll
  for (int i = 0; i < 4; i++) {
    v[i] = xr[tid + i * 256];
    s  += v[i];
    s2 += v[i] * v[i];
  }
  r1[tid] = s; r2[tid] = s2;
  __syncthreads();
  for (int off = 128; off > 0; off >>= 1) {
    if (tid < off) { r1[tid] += r1[tid + off]; r2[tid] += r2[tid + off]; }
    __syncthreads();
  }
  const float mu  = r1[0] * (1.f / Dm);
  const float var = r2[0] * (1.f / Dm) - mu * mu;
  const float rs  = rsqrtf(var + 1e-5f);
  u16* orow = out + (size_t)row * Dm;
#pragma unroll
  for (int i = 0; i < 4; i++) {
    int c = tid + i * 256;
    orow[c] = f2bf((v[i] - mu) * rs * g[c] + b[c]);
  }
}

// ---------------------------------------------------------------------------
// V transpose: vt[((b*H + h)*HD + d)*T + t] = v[(b*T + t)*D + h*64 + d]
// Coalesced along t on the write side; makes PV B-fragments contiguous b128s.
// ---------------------------------------------------------------------------
__global__ void transpose_v(const u16* __restrict__ vbf, u16* __restrict__ vt) {
  const size_t o0 = ((size_t)blockIdx.x * 256 + threadIdx.x) * 4;
  const int t0 = (int)(o0 & (Tq - 1));
  const int d  = (int)((o0 >> 10) & (HDh - 1));
  const int bh = (int)(o0 >> 16);
  const int b  = bh >> 4;
  const int h  = bh & (Hh - 1);
  const u16* src = vbf + (size_t)(b * Tq + t0) * Dm + h * HDh + d;
  u16* dst = vt + o0;
#pragma unroll
  for (int i = 0; i < 4; i++) dst[i] = src[(size_t)i * Dm];
}

// ---------------------------------------------------------------------------
// Staging helper: convert a 32x64 f32 weight tile -> bf16, transposed in LDS
// (dst[n_local*32 + k_local]) so B fragments become 2x ds_read_b128.
// ---------------------------------------------------------------------------
__device__ __forceinline__ void stage_b(u16* __restrict__ dst,
                                        const float* __restrict__ Bw,
                                        int k0, int N, int nBlk, int tid,
                                        bool prefetch_next) {
#pragma unroll
  for (int i = 0; i < 8; i++) {
    const int linear = (i * 128 + tid) * 2;   // over 32*64 elements
    const int kk = linear >> 6;
    const int nn = linear & 63;
    const float* src = Bw + (size_t)(k0 + kk) * N + nBlk + nn;
    f32x2 w = *(const f32x2*)src;
    dst[nn * 32 + kk]       = f2bf(w.x);
    dst[(nn + 1) * 32 + kk] = f2bf(w.y);
    if (prefetch_next) __builtin_prefetch(src + (size_t)32 * N, 0, 1);
  }
}

// ---------------------------------------------------------------------------
// GEMM: C[M,N] = A_bf16[M,K] * B_f32[K,N]  (+bias)(ReLU)(+resid) -> f32/bf16
// Block: 128 threads (4 waves), 64x64 C tile, wave w owns rows [w*16,w*16+16).
// Double-buffered LDS weight tiles; per chunk: read all fragments, issue 4
// back-to-back WMMAs, stage the next tile, single barrier.
// ---------------------------------------------------------------------------
__global__ void gemm_wmma(const u16* __restrict__ A, const float* __restrict__ Bw,
                          const float* __restrict__ bias,
                          const float* __restrict__ resid,
                          float* __restrict__ outF, u16* __restrict__ outB,
                          int N, int K, int relu) {
  __shared__ u16 bs[2][64 * 32];
  const int tid  = threadIdx.x;
  const int wave = tid >> 5;
  const int lane = tid & 31;
  const int half = lane >> 4;
  const int lm   = lane & 15;
  const int nBlk = blockIdx.x * 64;
  const int mBlk = blockIdx.y * 64;
  const int mRow = mBlk + wave * 16 + lm;     // A row for this lane

  v8f acc[4];
#pragma unroll
  for (int i = 0; i < 4; i++) acc[i] = v8f{0.f, 0.f, 0.f, 0.f, 0.f, 0.f, 0.f, 0.f};

  const int nIter = K >> 5;
  stage_b(bs[0], Bw, 0, N, nBlk, tid, nIter > 1);
  __syncthreads();

  for (int it = 0; it < nIter; it++) {
    const int k0 = it << 5;
    const u16* bcur = bs[it & 1];

    // A fragment: K = half*16 + 0..15 contiguous
    Frag af;
    const u16* ap = A + (size_t)mRow * K + k0 + half * 16;
    af.q[0] = *(const u32x4*)(ap);
    af.q[1] = *(const u32x4*)(ap + 8);

    // All four B fragments first (lane n = lm, K = half*16 + 0..15 contiguous)
    Frag bfr[4];
#pragma unroll
    for (int nt = 0; nt < 4; nt++) {
      const u16* bp = bcur + (nt * 16 + lm) * 32 + half * 16;
      bfr[nt].q[0] = *(const u32x4*)(bp);
      bfr[nt].q[1] = *(const u32x4*)(bp + 8);
    }

    // Back-to-back WMMAs
#pragma unroll
    for (int nt = 0; nt < 4; nt++)
      acc[nt] = wmma_bf16(af.v, bfr[nt].v, acc[nt]);

    // Stage next tile into the other buffer, then one barrier.
    if (it + 1 < nIter)
      stage_b(bs[(it + 1) & 1], Bw, k0 + 32, N, nBlk, tid, it + 2 < nIter);
    __syncthreads();
  }

  // epilogue: C layout — VGPR r holds (M = r + 8*half, N = lm)
#pragma unroll
  for (int nt = 0; nt < 4; nt++) {
    const int col = nBlk + nt * 16 + lm;
    const float bv = bias ? bias[col] : 0.f;
#pragma unroll
    for (int r = 0; r < 8; r++) {
      const int row = mBlk + wave * 16 + r + 8 * half;
      float v = acc[nt][r] + bv;
      if (relu) v = fmaxf(v, 0.f);
      const size_t off = (size_t)row * N + col;
      if (resid) v += resid[off];
      if (outF) outF[off] = v;
      if (outB) outB[off] = f2bf(v);
    }
  }
}

// ---------------------------------------------------------------------------
// Attention: one wave per (b, h, 16-query tile). Flash-style two passes:
// pass1 computes row maxima, pass2 recomputes S, forms unnormalized
// P = exp(S - mx), bounces P through 1KB LDS (C-layout -> A-layout), then PV
// via WMMA from the pre-transposed V; divide by row sums at the end.
// ---------------------------------------------------------------------------
__device__ __forceinline__ v8f qk_tile(const u16* __restrict__ kbf, int bT0,
                                       int kc, int hcol, int lm, int half,
                                       v16bf qa0, v16bf qa1) {
  // B operand = K matrix: N = key_local (lm), contraction = head dim.
  const u16* kr = kbf + (size_t)(bT0 + kc * 16 + lm) * Dm + hcol + half * 8;
  Frag k0f, k1f;
  k0f.q[0] = *(const u32x4*)(kr);
  k0f.q[1] = *(const u32x4*)(kr + 16);
  k1f.q[0] = *(const u32x4*)(kr + 32);
  k1f.q[1] = *(const u32x4*)(kr + 48);
  v8f s = v8f{0.f, 0.f, 0.f, 0.f, 0.f, 0.f, 0.f, 0.f};
  s = wmma_bf16(qa0, k0f.v, s);
  return wmma_bf16(qa1, k1f.v, s);
}

__global__ void attention_wmma(const u16* __restrict__ qbf,
                               const u16* __restrict__ kbf,
                               const u16* __restrict__ vt,
                               u16* __restrict__ attbf) {
  __shared__ u16 ps[16 * 32];   // P chunk (16 queries x 32 keys) bf16
  const int lane = threadIdx.x;
  const int half = lane >> 4;
  const int lm   = lane & 15;
  const int qt = blockIdx.x & 63;
  const int h  = (blockIdx.x >> 6) & 15;
  const int b  = blockIdx.x >> 10;
  const int bT0  = b * Tq;
  const int hcol = h * HDh;
  const float scale = 0.125f;   // HD^-0.5

  // Q fragments, 16x64 split into two K=32 chunks (kept in registers)
  Frag qa0, qa1;
  {
    const u16* qp = qbf + (size_t)(bT0 + qt * 16 + lm) * Dm + hcol;
    qa0.q[0] = *(const u32x4*)(qp + half * 8);
    qa0.q[1] = *(const u32x4*)(qp + 16 + half * 8);
    qa1.q[0] = *(const u32x4*)(qp + 32 + half * 8);
    qa1.q[1] = *(const u32x4*)(qp + 48 + half * 8);
  }

  float mx[8], sm[8];
#pragma unroll
  for (int r = 0; r < 8; r++) { mx[r] = -3.0e38f; sm[r] = 0.f; }

  // ---- pass 1: row maxima over all 1024 keys ----
  for (int kc = 0; kc < 64; kc++) {
    v8f s = qk_tile(kbf, bT0, kc, hcol, lm, half, qa0.v, qa1.v);
    const int skey = kc * 16 + lm;
#pragma unroll
    for (int r = 0; r < 8; r++) {
      const int t = qt * 16 + r + 8 * half;
      float v = is_masked(t, skey) ? -1.0e30f : s[r] * scale;
#pragma unroll
      for (int off = 1; off < 16; off <<= 1) v = fmaxf(v, __shfl_xor(v, off, 32));
      mx[r] = fmaxf(mx[r], v);
    }
  }

  // ---- pass 2: P = exp(S - mx); O += P @ V (unnormalized) ----
  v8f acc[4];
#pragma unroll
  for (int i = 0; i < 4; i++) acc[i] = v8f{0.f, 0.f, 0.f, 0.f, 0.f, 0.f, 0.f, 0.f};

  const u16* vhead = vt + (size_t)((b * Hh + h) * HDh) * Tq;

  for (int kc2 = 0; kc2 < 32; kc2++) {      // 32-key chunks
#pragma unroll
    for (int sub = 0; sub < 2; sub++) {
      const int kc = kc2 * 2 + sub;
      v8f s = qk_tile(kbf, bT0, kc, hcol, lm, half, qa0.v, qa1.v);
      const int skey = kc * 16 + lm;
#pragma unroll
      for (int r = 0; r < 8; r++) {
        const int t = qt * 16 + r + 8 * half;
        float v = is_masked(t, skey) ? -1.0e30f : s[r] * scale;
        float e = __expf(v - mx[r]);
        sm[r] += e;
        ps[(r + 8 * half) * 32 + sub * 16 + lm] = f2bf(e);
      }
    }
    __syncthreads();

    // A operand: P (rows = queries, contraction = 32 keys) from LDS
    Frag pa;
    const u16* pp = ps + lm * 32;
    pa.q[0] = *(const u32x4*)(pp + half * 16);
    pa.q[1] = *(const u32x4*)(pp + half * 16 + 8);

    // B operands: V from transposed layout; K-pairs along t are contiguous.
    Frag vb[4];
#pragma unroll
    for (int nt = 0; nt < 4; nt++) {
      const u16* vp = vhead + (size_t)(nt * 16 + lm) * Tq + kc2 * 32 + half * 16;
      vb[nt].q[0] = *(const u32x4*)(vp);
      vb[nt].q[1] = *(const u32x4*)(vp + 8);
    }
#pragma unroll
    for (int nt = 0; nt < 4; nt++)
      acc[nt] = wmma_bf16(pa.v, vb[nt].v, acc[nt]);
    __syncthreads();
  }

  // ---- epilogue: reduce row sums, normalize, store bf16 ----
#pragma unroll
  for (int r = 0; r < 8; r++) {
    float t = sm[r];
#pragma unroll
    for (int off = 1; off < 16; off <<= 1) t += __shfl_xor(t, off, 32);
    sm[r] = 1.f / t;
  }
#pragma unroll
  for (int nt = 0; nt < 4; nt++) {
#pragma unroll
    for (int r = 0; r < 8; r++) {
      const int row = bT0 + qt * 16 + r + 8 * half;
      attbf[(size_t)row * Dm + hcol + nt * 16 + lm] = f2bf(acc[nt][r] * sm[r]);
    }
  }
}

// ---------------------------------------------------------------------------
// Orchestration
// ---------------------------------------------------------------------------
extern "C" void kernel_launch(void* const* d_in, const int* in_sizes, int n_in,
                              void* d_out, int out_size, void* d_ws, size_t ws_size,
                              hipStream_t stream) {
  (void)in_sizes; (void)n_in; (void)out_size; (void)ws_size;
  const int*   idx     = (const int*)  d_in[0];
  const float* tok_emb = (const float*)d_in[1];
  const float* pos_emb = (const float*)d_in[2];
  const float* ln1_g   = (const float*)d_in[3];
  const float* ln1_b   = (const float*)d_in[4];
  const float* wq      = (const float*)d_in[5];
  const float* wk      = (const float*)d_in[6];
  const float* wv      = (const float*)d_in[7];
  const float* wproj   = (const float*)d_in[8];
  const float* bproj   = (const float*)d_in[9];
  const float* ln2_g   = (const float*)d_in[10];
  const float* ln2_b   = (const float*)d_in[11];
  const float* w1      = (const float*)d_in[12];
  const float* b1      = (const float*)d_in[13];
  const float* w2      = (const float*)d_in[14];
  const float* b2      = (const float*)d_in[15];
  const float* lnf_g   = (const float*)d_in[16];
  const float* lnf_b   = (const float*)d_in[17];
  const float* wlm     = (const float*)d_in[18];
  const float* blm     = (const float*)d_in[19];
  float* out = (float*)d_out;

  char* ws = (char*)d_ws;
  size_t off = 0;
  auto alloc = [&](size_t bytes) -> void* {
    void* p = ws + off;
    off += (bytes + 255) & ~(size_t)255;
    return p;
  };
  float* x   = (float*)alloc((size_t)Mrows * Dm * sizeof(float));
  u16* hbf   = (u16*)  alloc((size_t)Mrows * Dm * sizeof(u16));
  u16* qb    = (u16*)  alloc((size_t)Mrows * Dm * sizeof(u16));
  u16* kb    = (u16*)  alloc((size_t)Mrows * Dm * sizeof(u16));
  u16* vb    = (u16*)  alloc((size_t)Mrows * Dm * sizeof(u16));
  u16* vtb   = (u16*)  alloc((size_t)Mrows * Dm * sizeof(u16));
  u16* attb  = (u16*)  alloc((size_t)Mrows * Dm * sizeof(u16));
  u16* m1b   = (u16*)  alloc((size_t)Mrows * D4 * sizeof(u16));

  embed_kernel<<<Mrows, 256, 0, stream>>>(idx, tok_emb, pos_emb, x);

  const dim3 gD(Dm / 64, Mrows / 64);
  const dim3 g4(D4 / 64, Mrows / 64);
  for (int l = 0; l < Ll; l++) {
    layernorm_to_bf16<<<Mrows, 256, 0, stream>>>(x, ln1_g + l * Dm, ln1_b + l * Dm, hbf);
    gemm_wmma<<<gD, 128, 0, stream>>>(hbf, wq + (size_t)l * Dm * Dm, nullptr, nullptr,
                                      nullptr, qb, Dm, Dm, 0);
    gemm_wmma<<<gD, 128, 0, stream>>>(hbf, wk + (size_t)l * Dm * Dm, nullptr, nullptr,
                                      nullptr, kb, Dm, Dm, 0);
    gemm_wmma<<<gD, 128, 0, stream>>>(hbf, wv + (size_t)l * Dm * Dm, nullptr, nullptr,
                                      nullptr, vb, Dm, Dm, 0);
    transpose_v<<<(Mrows * Dm) / (256 * 4), 256, 0, stream>>>(vb, vtb);
    attention_wmma<<<2 * Hh * (Tq / 16), 32, 0, stream>>>(qb, kb, vtb, attb);
    gemm_wmma<<<gD, 128, 0, stream>>>(attb, wproj + (size_t)l * Dm * Dm, bproj + l * Dm,
                                      x, x, nullptr, Dm, Dm, 0);
    layernorm_to_bf16<<<Mrows, 256, 0, stream>>>(x, ln2_g + l * Dm, ln2_b + l * Dm, hbf);
    gemm_wmma<<<g4, 128, 0, stream>>>(hbf, w1 + (size_t)l * Dm * D4, b1 + (size_t)l * D4,
                                      nullptr, nullptr, m1b, D4, Dm, 1);
    gemm_wmma<<<gD, 128, 0, stream>>>(m1b, w2 + (size_t)l * D4 * Dm, b2 + l * Dm,
                                      x, x, nullptr, Dm, D4, 0);
  }
  layernorm_to_bf16<<<Mrows, 256, 0, stream>>>(x, lnf_g, lnf_b, hbf);
  gemm_wmma<<<dim3(NV2 / 64, Mrows / 64), 128, 0, stream>>>(hbf, wlm, blm, nullptr,
                                                            out, nullptr, NV2, Dm, 0);
}
